// Attention_73744588473066
// MI455X (gfx1250) — matrix operand order repaired and checked by reading, not compiled, not run on hardware
//
#include <hip/hip_runtime.h>
#include <hip/hip_bf16.h>
#include <math.h>

// ---------------------------------------------------------------------------
// CDNA5 / gfx1250 bf16 WMMA types & helpers
// ---------------------------------------------------------------------------
typedef __bf16 bf16;
typedef __bf16 v16bf __attribute__((ext_vector_type(16)));
typedef float  v8f   __attribute__((ext_vector_type(8)));

union BF16x16 { v16bf v; unsigned u[8]; bf16 h[16]; };

__device__ __forceinline__ v8f wmma_bf16(v16bf a, v16bf b, v8f c) {
    // D = A(16x32 bf16) * B(32x16 bf16) + C(16x16 f32)
    return __builtin_amdgcn_wmma_f32_16x16x32_bf16(
        false, a, false, b, (short)0, c, false, false);
}

// A fragment: lane holds row m = lane%16; K pairs depend on h = lane/16.
// p = this lane's row start at the fragment's K origin (32-wide slab).
__device__ __forceinline__ v16bf load_frag_a(const bf16* p, int h) {
    BF16x16 f;
#pragma unroll
    for (int j = 0; j < 8; ++j) {
        int K2 = h * 16 + 2 * j;
        f.u[j] = *(const unsigned*)(p + K2);
    }
    return f.v;
}

// B fragment from an N-major tile ([n][k] contiguous in k):
// lane holds col n = lane%16; p = lane's column start; K pairs h*16 + 2j.
__device__ __forceinline__ v16bf load_frag_b(const bf16* p, int h) {
    BF16x16 f;
#pragma unroll
    for (int j = 0; j < 8; ++j)
        f.u[j] = *(const unsigned*)(p + h * 16 + 2 * j);
    return f.v;
}

// B fragment from a K-major tile ([k][n], row stride ld): strided 16-bit reads.
// base = &tile[k_origin * ld + n].
__device__ __forceinline__ v16bf load_frag_b_kmajor(const bf16* base, int ld, int h) {
    BF16x16 f;
#pragma unroll
    for (int j = 0; j < 8; ++j) {
        int k = h * 16 + 2 * j;
        f.h[2 * j]     = base[(size_t)k * ld];
        f.h[2 * j + 1] = base[(size_t)(k + 1) * ld];
    }
    return f.v;
}

// Async global->LDS copy (CDNA5): 16B per lane, tracked with ASYNCcnt.
#define LDS_OFF(p) ((unsigned)(unsigned long long)(p))
__device__ __forceinline__ void async_copy_b128(unsigned lds_off, const bf16* g) {
    asm volatile("global_load_async_to_lds_b128 %0, %1, off"
                 :: "v"(lds_off), "v"((unsigned long long)g)
                 : "memory");
}

// ---------------------------------------------------------------------------
// Tiled GEMM: C[M,Nout] = A[M,K](f32->bf16) @ W[K,Nout](f32->bf16)
// Block 128x128, 8 waves, each wave 32x64 (8 WMMAs / K-step).
// MODE 0: store bf16 (value * oscale)
// MODE 1: store f32 sigmoid(value + bias[col])   (gate path)
// MODE 2: store f32 value                        (output projection)
// ---------------------------------------------------------------------------
#define GBM 128
#define GBN 128
#define GBK 32
#define GLDA 34   // even; 17-bank row stride (conflict-free)
#define GLDB 34

template <int MODE>
__global__ __launch_bounds__(256)
void gemm_bf16_kernel(const float* __restrict__ A, const float* __restrict__ W,
                      const float* __restrict__ bias, void* __restrict__ out,
                      int M, int K, int Nout, float oscale)
{
    __shared__ bf16 As[GBM * GLDA];          // [m][k]
    __shared__ bf16 Bs[GBN * GLDB];          // [n][k] (transposed)

    const int tid  = threadIdx.x;
    const int lane = tid & 31;
    const int wave = tid >> 5;
    const int wm   = wave >> 1;              // 0..3 : 32-row slab
    const int wn   = wave & 1;               // 0..1 : 64-col slab
    const int lm   = lane & 15;
    const int lh   = lane >> 4;
    const int m0   = blockIdx.y * GBM;
    const int n0   = blockIdx.x * GBN;

    v8f acc[2][4] = {};

    for (int k0 = 0; k0 < K; k0 += GBK) {
        __syncthreads();
        // Stage A tile (128x32 f32 -> bf16)
#pragma unroll
        for (int it = 0; it < (GBM * GBK) / 256; ++it) {
            int idx = tid + it * 256;
            int r = idx >> 5, c = idx & 31;
            As[r * GLDA + c] = (bf16)A[(size_t)(m0 + r) * K + k0 + c];
        }
        // Stage W tile transposed (32x128 f32 -> bf16, stored [n][k])
#pragma unroll
        for (int it = 0; it < (GBK * GBN) / 256; ++it) {
            int idx = tid + it * 256;
            int r = idx >> 7, c = idx & 127;
            Bs[c * GLDB + r] = (bf16)W[(size_t)(k0 + r) * Nout + n0 + c];
        }
        __syncthreads();

        v16bf a0 = load_frag_a(&As[(wm * 32 + lm) * GLDA], lh);
        v16bf a1 = load_frag_a(&As[(wm * 32 + 16 + lm) * GLDA], lh);
#pragma unroll
        for (int ni = 0; ni < 4; ++ni) {
            v16bf bb = load_frag_b(&Bs[(wn * 64 + ni * 16 + lm) * GLDB], lh);
            acc[0][ni] = wmma_bf16(a0, bb, acc[0][ni]);
            acc[1][ni] = wmma_bf16(a1, bb, acc[1][ni]);
        }
    }

    // Epilogue: C layout row = (lane/16)*8 + r, col = lane%16
#pragma unroll
    for (int mi = 0; mi < 2; ++mi) {
#pragma unroll
        for (int ni = 0; ni < 4; ++ni) {
            int rbase = m0 + wm * 32 + mi * 16 + lh * 8;
            int col   = n0 + wn * 64 + ni * 16 + lm;
#pragma unroll
            for (int r = 0; r < 8; ++r) {
                float v = acc[mi][ni][r];
                size_t o = (size_t)(rbase + r) * Nout + col;
                if (MODE == 0) {
                    ((bf16*)out)[o] = (bf16)(v * oscale);
                } else if (MODE == 1) {
                    ((float*)out)[o] =
                        __builtin_amdgcn_rcpf(1.0f + __expf(-(v + bias[col])));
                } else {
                    ((float*)out)[o] = v;
                }
            }
        }
    }
}

// ---------------------------------------------------------------------------
// Flash attention with fused bias/mask/gate.
// Block: 256 threads (8 waves), 128 query rows (16 per wave), key tiles of 64.
// KV tiles staged by double-buffered GLOBAL_LOAD_ASYNC_TO_LDS_B128 (ASYNCcnt).
// q is pre-scaled by DH^-0.5 (folded into the bf16 store of kernel 1).
// ---------------------------------------------------------------------------
#define A_N   2048
#define A_H   16
#define A_DH  64
#define A_DI  1024
#define LDKS  88   // row stride: 176B (16B aligned for B128 async), 44-bank step
#define LDP   66

__global__ __launch_bounds__(256)
void attn_kernel(const bf16* __restrict__ qbuf,   // [B*N, DI] bf16, pre-scaled
                 const bf16* __restrict__ kvbuf,  // [B*N, 2*DI] bf16 (k | v)
                 const float* __restrict__ bias,  // [B,H,N,N] f32
                 const unsigned char* __restrict__ mask,   // [B,N] bool
                 const float* __restrict__ gates, // [B*N, DI] f32
                 float* __restrict__ att)         // [B*N, DI] f32
{
    __shared__ bf16 Kb[2][64 * LDKS];        // [j][d] row-major (async dest)
    __shared__ bf16 Vb[2][64 * LDKS];        // [j][d] row-major (async dest)
    __shared__ bf16 Ps[8 * 16 * LDP];        // per-wave 16x64 P slab

    const int tid  = threadIdx.x;
    const int lane = tid & 31;
    const int wave = tid >> 5;
    const int lm   = lane & 15;
    const int lh   = lane >> 4;
    const int b    = blockIdx.z;
    const int h    = blockIdx.y;
    const int i0   = blockIdx.x * 128 + wave * 16;   // this wave's 16 query rows

    // Load the wave's Q fragments once (16 rows x 64 d -> 2 A fragments)
    const bf16* qrow = qbuf + (size_t)(b * A_N + i0 + lm) * A_DI + h * A_DH;
    v16bf qf0 = load_frag_a(qrow, lh);
    v16bf qf1 = load_frag_a(qrow + 32, lh);

    float rmax[8], rsum[8];
#pragma unroll
    for (int r = 0; r < 8; ++r) { rmax[r] = -3.0e38f; rsum[r] = 0.0f; }
    v8f oacc[4] = {};

    bf16* myP = Ps + wave * 16 * LDP;
    const float* biasrow = bias + ((size_t)(b * A_H + h) * A_N + (i0 + lh * 8)) * A_N;

    // Issue the async KV copies for key tile t into stage t&1 (4 ops/thread).
    auto issue_tile = [&](int t) {
        int stage = t & 1;
        int j0t = t * 64;
#pragma unroll
        for (int hc = 0; hc < 2; ++hc) {
            int ch = tid + hc * 256;                 // 512 chunks of 16B
            int j  = ch >> 3;
            int c8 = (ch & 7) * 8;                   // element offset in row
            const bf16* g = kvbuf + (size_t)(b * A_N + j0t + j) * (2 * A_DI)
                                  + h * A_DH + c8;
            async_copy_b128(LDS_OFF(&Kb[stage][j * LDKS + c8]), g);
            async_copy_b128(LDS_OFF(&Vb[stage][j * LDKS + c8]), g + A_DI);
        }
    };

    issue_tile(0);

    for (int t = 0; t < A_N / 64; ++t) {
        const int j0 = t * 64;
        if (t + 1 < A_N / 64) {
            issue_tile(t + 1);                        // prefetch next stage
            __builtin_prefetch(biasrow + j0 + 64 + lm * 4, 0, 0);  // next bias tile
            asm volatile("s_wait_asynccnt 0x4" ::: "memory");  // tile t landed
        } else {
            asm volatile("s_wait_asynccnt 0x0" ::: "memory");
        }
        __syncthreads();                              // all lanes' copies visible

        const bf16* Ks = Kb[t & 1];
        const bf16* Vs = Vb[t & 1];

        // S = Q @ K^T : 4 col-tiles of 16, d-dim 64 in two 32 steps.
        // Ks is [j][d]: for the B operand (d x j) that is the N-major layout.
        v8f sacc[4] = {};
#pragma unroll
        for (int jt = 0; jt < 4; ++jt) {
            v16bf bk0 = load_frag_b(&Ks[(jt * 16 + lm) * LDKS], lh);
            v16bf bk1 = load_frag_b(&Ks[(jt * 16 + lm) * LDKS + 32], lh);
            sacc[jt] = wmma_bf16(qf0, bk0, sacc[jt]);
            sacc[jt] = wmma_bf16(qf1, bk1, sacc[jt]);
        }

        // Fuse attn_bias (streamed once from HBM) and the key mask.
#pragma unroll
        for (int jt = 0; jt < 4; ++jt) {
            int j = j0 + jt * 16 + lm;
            bool keep = mask[b * A_N + j] != 0;
            const float* bp = biasrow + j;
#pragma unroll
            for (int r = 0; r < 8; ++r) {
                float s = sacc[jt][r] + bp[(size_t)r * A_N];
                sacc[jt][r] = keep ? s : -3.0e38f;
            }
        }

        // Online softmax (rows live in 16-lane half-waves) + P to LDS (bf16).
#pragma unroll
        for (int r = 0; r < 8; ++r) {
            float mloc = fmaxf(fmaxf(sacc[0][r], sacc[1][r]),
                               fmaxf(sacc[2][r], sacc[3][r]));
#pragma unroll
            for (int off = 8; off >= 1; off >>= 1)
                mloc = fmaxf(mloc, __shfl_xor(mloc, off, 32));
            float mnew  = fmaxf(rmax[r], mloc);
            float alpha = __expf(rmax[r] - mnew);
            rmax[r] = mnew;
            float psum = 0.0f;
#pragma unroll
            for (int jt = 0; jt < 4; ++jt) {
                float p = __expf(sacc[jt][r] - mnew);
                psum += p;
                myP[(lh * 8 + r) * LDP + jt * 16 + lm] = (bf16)p;
            }
#pragma unroll
            for (int off = 8; off >= 1; off >>= 1)
                psum += __shfl_xor(psum, off, 32);
            rsum[r] = rsum[r] * alpha + psum;
#pragma unroll
            for (int dt = 0; dt < 4; ++dt)
                oacc[dt][r] *= alpha;
        }

        // O += P @ V.  P re-read from per-wave LDS as A fragments;
        // Vs is [j][d] = K-major for the B operand -> strided 16-bit reads.
#pragma unroll
        for (int kk = 0; kk < 2; ++kk) {
            v16bf pa = load_frag_a(myP + lm * LDP + kk * 32, lh);
#pragma unroll
            for (int dt = 0; dt < 4; ++dt) {
                v16bf vb = load_frag_b_kmajor(&Vs[(kk * 32) * LDKS + dt * 16 + lm],
                                              LDKS, lh);
                oacc[dt] = wmma_bf16(pa, vb, oacc[dt]);
            }
        }
        __syncthreads();          // done reading stage t&1 before its reuse
    }

    // Normalize (v_rcp) and fuse the sigmoid-gate multiply.
#pragma unroll
    for (int r = 0; r < 8; ++r) {
        float inv = __builtin_amdgcn_rcpf(rsum[r]);
        int row = i0 + lh * 8 + r;
#pragma unroll
        for (int dt = 0; dt < 4; ++dt) {
            int col = h * A_DH + dt * 16 + lm;
            size_t o = (size_t)(b * A_N + row) * A_DI + col;
            att[o] = (oacc[dt][r] * inv) * gates[o];
        }
    }
}

// ---------------------------------------------------------------------------
// Host launcher
// ---------------------------------------------------------------------------
extern "C" void kernel_launch(void* const* d_in, const int* in_sizes, int n_in,
                              void* d_out, int out_size, void* d_ws, size_t ws_size,
                              hipStream_t stream) {
    (void)in_sizes; (void)n_in; (void)out_size; (void)ws_size;
    const float*         seq  = (const float*)d_in[0];
    const unsigned char* mask = (const unsigned char*)d_in[1];   // bool [B,N]
    const float*         bias = (const float*)d_in[2];           // [B,H,N,N]
    const float*         Wq   = (const float*)d_in[3];
    const float*         Wkv  = (const float*)d_in[4];
    const float*         Wo   = (const float*)d_in[5];
    const float*         Wg   = (const float*)d_in[6];
    const float*         bg   = (const float*)d_in[7];
    float*               out  = (float*)d_out;

    const int B = 2, N = 2048, D = 1024, DI = 1024;
    const int M = B * N;                       // 4096 combined rows
    const float scale = 0.125f;                // 64^-0.5

    // Workspace layout (56 MB):
    char* ws = (char*)d_ws;
    bf16*  qbuf  = (bf16*)(ws);                          //  8 MB  [M,DI] bf16
    bf16*  kvbuf = (bf16*)(ws + ((size_t)8  << 20));     // 16 MB  [M,2DI] bf16
    float* gates = (float*)(ws + ((size_t)24 << 20));    // 16 MB  [M,DI] f32
    float* att   = (float*)(ws + ((size_t)40 << 20));    // 16 MB  [M,DI] f32

    dim3 blk(256);
    // q = (seq @ Wq) * scale -> bf16 (scale folded into store)
    gemm_bf16_kernel<0><<<dim3(DI / GBN, M / GBM), blk, 0, stream>>>(
        seq, Wq, nullptr, qbuf, M, D, DI, scale);
    // kv = seq @ Wkv -> bf16
    gemm_bf16_kernel<0><<<dim3((2 * DI) / GBN, M / GBM), blk, 0, stream>>>(
        seq, Wkv, nullptr, kvbuf, M, D, 2 * DI, 1.0f);
    // gates = sigmoid(seq @ Wg + bg) -> f32
    gemm_bf16_kernel<1><<<dim3(DI / GBN, M / GBM), blk, 0, stream>>>(
        seq, Wg, bg, gates, M, D, DI, 1.0f);
    // fused flash attention + bias + mask + gate (async double-buffered KV)
    attn_kernel<<<dim3(N / 128, 16, B), blk, 0, stream>>>(
        qbuf, kvbuf, bias, mask, gates, att);
    // out = att @ Wo -> f32
    gemm_bf16_kernel<2><<<dim3(D / GBN, M / GBM), blk, 0, stream>>>(
        att, Wo, nullptr, out, M, DI, D, 1.0f);
}
